// PairformerModule_21466246545422
// MI455X (gfx1250) — compile-verified, matrix operand order checked by reading, not run on hardware
//
#include <hip/hip_runtime.h>

// ============================================================================
// Pairformer (2 layers) for MI455X / gfx1250 (wave32, WMMA).
// All large GEMMs (>=97% of the ~200 GFLOP) run through
// v_wmma_f32_16x16x32_f16 (f16 operands converted on the fly, f32 accum).
// z tensor (33.5MB) is L2-resident on MI455X (192MB L2) -> compute bound.
// GEMM: templated NN/NT/TN variants (no runtime branches in inner loop),
// 64x64 block tile, 8 waves x (16x32) per wave, LDS-staged f16 operands with
// B stored transposed so every WMMA fragment is two contiguous ds_load_b128.
//
// Param leaf order assumption: recursive dict-INSERTION-order flattening:
//   top: s, z, mask, pair_mask, params[0..1]
//   layer: tri_out(8) tri_in(8) att_start(8) att_end(8) trans_z(5) attn(9)
//          sln_g sln_b trans_s(5)   => 53 leaves / layer
// Workspace use: ~510 MB of d_ws (fp32 scratch buffers, laid out below).
// ============================================================================

#define NT   256
#define P2   65536
#define TZc  128
#define TSc  384

typedef __attribute__((ext_vector_type(16))) _Float16 v16h;
typedef __attribute__((ext_vector_type(8)))  _Float16 v8h;
typedef __attribute__((ext_vector_type(8)))  float    v8f;
typedef __attribute__((ext_vector_type(4)))  float    v4f;

__device__ __forceinline__ float sigf(float x){ return 1.0f/(1.0f + __expf(-x)); }

__device__ __forceinline__ v8f wmma16(v16h a, v16h b, v8f c){
  return __builtin_amdgcn_wmma_f32_16x16x32_f16(false, a, false, b, (short)0, c, false, false);
}
__device__ __forceinline__ v16h cat8(v8h lo, v8h hi){
  return __builtin_shufflevector(lo, hi, 0,1,2,3,4,5,6,7,8,9,10,11,12,13,14,15);
}
// convert 8 contiguous f32 (16B-aligned) -> v8h via two b128 loads
__device__ __forceinline__ v8h cvt8(const float* __restrict__ p){
  v4f a = *(const v4f*)p;
  v4f b = *(const v4f*)(p + 4);
  v8h r;
  r[0]=(_Float16)a[0]; r[1]=(_Float16)a[1]; r[2]=(_Float16)a[2]; r[3]=(_Float16)a[3];
  r[4]=(_Float16)b[0]; r[5]=(_Float16)b[1]; r[6]=(_Float16)b[2]; r[7]=(_Float16)b[3];
  return r;
}

// ------------------------------- LayerNorm ---------------------------------
__global__ void k_ln(const float* __restrict__ x, const float* __restrict__ g,
                     const float* __restrict__ b, float* __restrict__ y, int D){
  int row = blockIdx.x;
  const float* xr = x + (long)row * D;
  float* yr = y + (long)row * D;
  __shared__ float red[128];
  int t = threadIdx.x;
  float s = 0.f;
  for (int i = t; i < D; i += 128) s += xr[i];
  red[t] = s; __syncthreads();
  for (int o = 64; o > 0; o >>= 1){ if (t < o) red[t] += red[t+o]; __syncthreads(); }
  float mu = red[0] / (float)D; __syncthreads();
  float v = 0.f;
  for (int i = t; i < D; i += 128){ float d = xr[i]-mu; v += d*d; }
  red[t] = v; __syncthreads();
  for (int o = 64; o > 0; o >>= 1){ if (t < o) red[t] += red[t+o]; __syncthreads(); }
  float rstd = rsqrtf(red[0]/(float)D + 1e-5f); __syncthreads();
  for (int i = t; i < D; i += 128) yr[i] = (xr[i]-mu)*rstd*g[i] + b[i];
}

// ------------------------- Generic WMMA GEMM -------------------------------
// C[M,N] = A_log[M,K] * B_log[K,N]
// TA: A phys [K][M] ; TB: B phys [N][K]. Compile-time -> straight-line loads.
// Block 256 thr (8 waves), tile 64x64, K-step 32; wave tile 16x32 (2 accum).
template<int TA, int TB>
__global__ __launch_bounds__(256)
void k_gemm_t(const float* __restrict__ A, const float* __restrict__ B,
              float* __restrict__ C, int M, int Nn, int K,
              long bA, long bB, long bC){
  const float* Ab = A + bA * blockIdx.z;
  const float* Bb = B + bB * blockIdx.z;
  float* Cb = C + bC * blockIdx.z;
  int m0 = blockIdx.y * 64, n0 = blockIdx.x * 64;
  __shared__ _Float16 sa [64][40];   // [m][k]
  __shared__ _Float16 sbt[64][40];   // [n][k]  (B transposed in LDS)
  int tid = threadIdx.x, lane = tid & 31, wave = tid >> 5;
  int wm = wave & 3, wn = wave >> 2;
  int hrow = lane & 15, koff = (lane >> 4) * 8;
  v8f acc0 = {}, acc1 = {};
  for (int k0 = 0; k0 < K; k0 += 32){
    if (TA){
      for (int idx = tid; idx < 2048; idx += 256){
        int m = idx & 63, kk = idx >> 6;               // coalesce over m
        sa[m][kk] = (_Float16)Ab[(long)(k0+kk)*M + m0 + m];
      }
    } else {
      for (int idx = tid; idx < 2048; idx += 256){
        int m = idx >> 5, kk = idx & 31;               // coalesce over k
        sa[m][kk] = (_Float16)Ab[(long)(m0+m)*K + k0 + kk];
      }
    }
    if (TB){
      for (int idx = tid; idx < 2048; idx += 256){
        int n = idx >> 5, kk = idx & 31;               // coalesce over k
        sbt[n][kk] = (_Float16)Bb[(long)(n0+n)*K + k0 + kk];
      }
    } else {
      for (int idx = tid; idx < 2048; idx += 256){
        int n = idx & 63, kk = idx >> 6;               // coalesce over n
        sbt[n][kk] = (_Float16)Bb[(long)(k0+kk)*Nn + n0 + n];
      }
    }
    __syncthreads();
    int ar = wm*16 + hrow;
    v16h a  = cat8(*(const v8h*)&sa[ar][koff],  *(const v8h*)&sa[ar][16+koff]);
    int bc0 = wn*32 + hrow, bc1 = bc0 + 16;
    v16h b0 = cat8(*(const v8h*)&sbt[bc0][koff], *(const v8h*)&sbt[bc0][16+koff]);
    v16h b1 = cat8(*(const v8h*)&sbt[bc1][koff], *(const v8h*)&sbt[bc1][16+koff]);
    acc0 = wmma16(a, b0, acc0);
    acc1 = wmma16(a, b1, acc1);
    __syncthreads();
  }
  int nA = n0 + wn*32 + (lane & 15);
  int mb = m0 + wm*16 + ((lane >> 4) ? 8 : 0);
#pragma unroll
  for (int r = 0; r < 8; ++r){
    Cb[(long)(mb + r)*Nn + nA]      = acc0[r];
    Cb[(long)(mb + r)*Nn + nA + 16] = acc1[r];
  }
}

// -------------------- small-N GEMM (Nout = 4 or 16) ------------------------
__global__ void k_gemm_small(const float* __restrict__ A, const float* __restrict__ B,
                             float* __restrict__ C, int M, int Nn, int K, int colmajor){
  long idx = (long)blockIdx.x*256 + threadIdx.x;
  if (idx >= (long)M*Nn) return;
  int m = (int)(idx / Nn), n = (int)(idx % Nn);
  float s = 0.f;
  for (int k = 0; k < K; ++k) s += A[(long)m*K + k]*B[(long)k*Nn + n];
  if (colmajor) C[(long)n*M + m] = s; else C[(long)m*Nn + n] = s;
}

// --------------------------- elementwise -----------------------------------
__global__ void k_glu_mask(const float* P, const float* G, const float* pm,
                           float* O, int Cc, long total){
  long i = (long)blockIdx.x*256 + threadIdx.x; if (i >= total) return;
  long r = i / Cc;
  O[i] = P[i]*sigf(G[i])*pm[r];
}
__global__ void k_mul_sig(const float* A, const float* B, float* O, long total){
  long i = (long)blockIdx.x*256 + threadIdx.x; if (i >= total) return;
  O[i] = A[i]*sigf(B[i]);
}
__global__ void k_gate_add(float* D, const float* U, const float* G, long total){
  long i = (long)blockIdx.x*256 + threadIdx.x; if (i >= total) return;
  D[i] += U[i]*sigf(G[i]);
}
__global__ void k_silu_mul(const float* A, const float* B, float* O, long total){
  long i = (long)blockIdx.x*256 + threadIdx.x; if (i >= total) return;
  float x = A[i];
  O[i] = x*sigf(x)*B[i];
}
__global__ void k_add(float* D, const float* S, long total){
  long i = (long)blockIdx.x*256 + threadIdx.x; if (i >= total) return;
  D[i] += S[i];
}
__global__ void k_add_bias(float* X, const float* b, int Cc, long total){
  long i = (long)blockIdx.x*256 + threadIdx.x; if (i >= total) return;
  X[i] += b[i % Cc];
}

// ----------------------------- transposes ----------------------------------
__global__ void k_transpose_ij(const float* __restrict__ S, float* __restrict__ D, int Dd, long total){
  long idx = (long)blockIdx.x*256 + threadIdx.x; if (idx >= total) return;
  long pq = idx / Dd; int d = (int)(idx % Dd);
  int i = (int)(pq >> 8), j = (int)(pq & 255);
  D[((long)((j<<8)|i))*Dd + d] = S[idx];
}
// [pq, Ctot] channel slice -> [d][pq]
__global__ void k_split_dpq(const float* __restrict__ S, float* __restrict__ D, int chOff, int Ctot){
  long idx = (long)blockIdx.x*256 + threadIdx.x;
  if (idx >= (long)P2*128) return;
  int d = (int)(idx & 127); long pq = idx >> 7;
  D[(long)d*P2 + pq] = S[pq*Ctot + chOff + d];
}
__global__ void k_from_dpq(const float* __restrict__ S, float* __restrict__ D){
  long idx = (long)blockIdx.x*256 + threadIdx.x;
  if (idx >= (long)P2*128) return;
  int d = (int)(idx & 127); long pq = idx >> 7;
  D[pq*128 + d] = S[(long)d*P2 + pq];
}

// ------------------- triangle attention core (WMMA) ------------------------
// one wave per (q-tile of 16, head h in 0..3, row i). Q/K/V: [P2][128] (ch = h*32+c)
__global__ __launch_bounds__(32)
void k_tri_attn(const float* __restrict__ Q, const float* __restrict__ Kx,
                const float* __restrict__ V, const float* __restrict__ TB,
                const float* __restrict__ pm, float* __restrict__ O){
  int qt = blockIdx.x, h = blockIdx.y, i = blockIdx.z;
  int lane = threadIdx.x;
  __shared__ float    sc[16][256];
  __shared__ _Float16 sp[16][256];
  __shared__ _Float16 vt[32][40];     // [c][k] chunk of V, transposed
  int hrow = lane & 15, koff = (lane >> 4) * 8;
  int q0 = qt * 16;
  const float scale = 0.17677669529663687f; // 1/sqrt(32)
  // scores = scale*(Q K^T) + tri_bias + mask_bias
  const float* qb = Q + ((long)(i*256 + q0 + hrow))*128 + h*32 + koff;
  v16h a = cat8(cvt8(qb), cvt8(qb + 16));
  for (int kt = 0; kt < 16; ++kt){
    const float* kb = Kx + ((long)(i*256 + kt*16 + hrow))*128 + h*32 + koff;
    v16h b = cat8(cvt8(kb), cvt8(kb + 16));
    v8f acc = {};
    acc = wmma16(a, b, acc);
    int n = kt*16 + (lane & 15);
    int rb = (lane >> 4) * 8;
    float mb = 1e9f*(pm[i*256 + n] - 1.0f);
#pragma unroll
    for (int r = 0; r < 8; ++r){
      int qr = rb + r;
      sc[qr][n] = acc[r]*scale + TB[(long)h*P2 + (long)(q0+qr)*256 + n] + mb;
    }
  }
  __syncthreads();
  if (lane < 16){
    float mx = -1e30f;
    for (int k2 = 0; k2 < 256; ++k2) mx = fmaxf(mx, sc[lane][k2]);
    float sum = 0.f;
    for (int k2 = 0; k2 < 256; ++k2){ float e = __expf(sc[lane][k2]-mx); sc[lane][k2] = e; sum += e; }
    float inv = 1.0f/sum;
    for (int k2 = 0; k2 < 256; ++k2) sp[lane][k2] = (_Float16)(sc[lane][k2]*inv);
  }
  __syncthreads();
  // O(16x32) = A(16x256) V(256x32); V chunk staged transposed in LDS
  v8f o0 = {}, o1 = {};
  for (int k0 = 0; k0 < 256; k0 += 32){
    for (int idx = lane; idx < 1024; idx += 32){
      int kk = idx >> 5, c = idx & 31;                 // coalesce over c
      vt[c][kk] = (_Float16)V[((long)(i*256 + k0 + kk))*128 + h*32 + c];
    }
    __syncthreads();
    v16h ap = cat8(*(const v8h*)&sp[hrow][k0+koff],  *(const v8h*)&sp[hrow][k0+16+koff]);
    v16h b0 = cat8(*(const v8h*)&vt[hrow][koff],     *(const v8h*)&vt[hrow][16+koff]);
    v16h b1 = cat8(*(const v8h*)&vt[16+hrow][koff],  *(const v8h*)&vt[16+hrow][16+koff]);
    o0 = wmma16(ap, b0, o0);
    o1 = wmma16(ap, b1, o1);
    __syncthreads();
  }
  int rb = (lane >> 4) * 8;
#pragma unroll
  for (int r = 0; r < 8; ++r){
    float* ob = O + ((long)(i*256 + q0 + rb + r))*128 + h*32;
    ob[lane & 15]        = o0[r];
    ob[16 + (lane & 15)] = o1[r];
  }
}

// -------------------- single-seq attention (dh=24, VALU) -------------------
__global__ __launch_bounds__(256)
void k_s_attn(const float* __restrict__ Qs, const float* __restrict__ Ks,
              const float* __restrict__ Vs, const float* __restrict__ ZB,
              const float* __restrict__ mask, float* __restrict__ Os){
  int i = blockIdx.x, h = blockIdx.y, j = threadIdx.x;
  __shared__ float sc[256];
  __shared__ float red[256];
  const float scale = 0.2041241452319315f; // 1/sqrt(24)
  float s = 0.f;
  for (int d = 0; d < 24; ++d) s += Qs[i*TSc + h*24 + d]*Ks[j*TSc + h*24 + d];
  float v = s*scale + ZB[((long)i*256 + j)*16 + h] - 1e9f*(1.0f - mask[j]);
  red[j] = v; __syncthreads();
  for (int o = 128; o > 0; o >>= 1){ if (j < o) red[j] = fmaxf(red[j], red[j+o]); __syncthreads(); }
  float mx = red[0]; __syncthreads();
  float e = __expf(v - mx);
  sc[j] = e; red[j] = e; __syncthreads();
  for (int o = 128; o > 0; o >>= 1){ if (j < o) red[j] += red[j+o]; __syncthreads(); }
  float inv = 1.0f/red[0]; __syncthreads();
  sc[j] = e*inv; __syncthreads();
  if (j < 24){
    float o = 0.f;
    for (int jj = 0; jj < 256; ++jj) o += sc[jj]*Vs[jj*TSc + h*24 + j];
    Os[i*TSc + h*24 + j] = o;
  }
}

// ============================ host orchestration ============================
static inline int ew_blocks(long total){ return (int)((total + 255) / 256); }

extern "C" void kernel_launch(void* const* d_in, const int* in_sizes, int n_in,
                              void* d_out, int out_size, void* d_ws, size_t ws_size,
                              hipStream_t stream) {
  (void)in_sizes; (void)n_in; (void)out_size; (void)ws_size;
  const float* pm   = (const float*)d_in[3];   // pair_mask [P2]
  const float* msk  = (const float*)d_in[2];   // mask [256]

  float* s = (float*)d_out;                    // [256*384]
  float* z = s + 256*TSc;                      // [P2*128]
  hipMemcpyAsync(s, d_in[0], (size_t)256*TSc*sizeof(float), hipMemcpyDeviceToDevice, stream);
  hipMemcpyAsync(z, d_in[1], (size_t)P2*128*sizeof(float), hipMemcpyDeviceToDevice, stream);

  // workspace layout (fp32)
  float* w = (float*)d_ws;
  size_t off = 0;
  auto alloc = [&](size_t n){ float* p = w + off; off += n; return p; };
  float* ZN   = alloc((size_t)P2*128);
  float* BIG0 = alloc((size_t)P2*512);
  float* BIG1 = alloc((size_t)P2*512);
  float* DA   = alloc((size_t)P2*128);
  float* DB   = alloc((size_t)P2*128);
  float* DT   = alloc((size_t)P2*128);
  float* T0   = alloc((size_t)P2*128);
  float* T1   = alloc((size_t)P2*128);
  float* T2   = alloc((size_t)P2*128);
  float* ZT   = alloc((size_t)P2*128);
  float* PMT  = alloc((size_t)P2);
  float* TBB  = alloc((size_t)4*P2);
  float* ZBB  = alloc((size_t)P2*16);
  float* SN   = alloc((size_t)256*TSc);
  float* SQ   = alloc((size_t)256*TSc);
  float* SK   = alloc((size_t)256*TSc);
  float* SV   = alloc((size_t)256*TSc);
  float* SG   = alloc((size_t)256*TSc);
  float* SO   = alloc((size_t)256*TSc);
  float* SO2  = alloc((size_t)256*TSc);
  float* SH1  = alloc((size_t)256*1536);
  float* SH2  = alloc((size_t)256*1536);

  auto gemm = [&](const float* A, const float* B, float* C, int M, int Nn, int K,
                  int tA = 0, int tB = 0, long bA = 0, long bB = 0, long bC = 0, int batch = 1){
    dim3 g(Nn/64, M/64, batch);
    if (tA == 0 && tB == 0)      k_gemm_t<0,0><<<g, 256, 0, stream>>>(A, B, C, M, Nn, K, bA, bB, bC);
    else if (tA == 0 && tB == 1) k_gemm_t<0,1><<<g, 256, 0, stream>>>(A, B, C, M, Nn, K, bA, bB, bC);
    else                         k_gemm_t<1,0><<<g, 256, 0, stream>>>(A, B, C, M, Nn, K, bA, bB, bC);
  };

  for (int layer = 0; layer < 2; ++layer){
    const float* const* L = (const float* const*)(d_in + 4 + layer*53);
    // leaf offsets (insertion order)
    const int TRI_OUT = 0, TRI_IN = 8, ATT_START = 16, ATT_END = 24;
    const int TRANS_Z = 32, ATTN = 37, SLN_G = 46, SLN_B = 47, TRANS_S = 48;

    // ---- triangle multiplicative update (outgoing then incoming) ----
    auto tri_mult = [&](int b, int outgoing){
      const float *ln_in_g = L[b+0], *ln_in_b = L[b+1], *p_in = L[b+2], *g_in = L[b+3];
      const float *ln_out_g = L[b+4], *ln_out_b = L[b+5], *p_out = L[b+6], *g_out = L[b+7];
      k_ln<<<P2, 128, 0, stream>>>(z, ln_in_g, ln_in_b, ZN, 128);
      gemm(ZN, p_in, BIG0, P2, 256, 128);
      gemm(ZN, g_in, BIG1, P2, 256, 128);
      k_glu_mask<<<ew_blocks((long)P2*256), 256, 0, stream>>>(BIG0, BIG1, pm, BIG0, 256, (long)P2*256);
      k_split_dpq<<<ew_blocks((long)P2*128), 256, 0, stream>>>(BIG0, DA, 0,   256);
      k_split_dpq<<<ew_blocks((long)P2*128), 256, 0, stream>>>(BIG0, DB, 128, 256);
      // outgoing: C_d = A_d * B_d^T ; incoming: C_d = A_d^T * B_d
      gemm(DA, DB, DT, 256, 256, 256, outgoing ? 0 : 1, outgoing ? 1 : 0,
           (long)P2, (long)P2, (long)P2, 128);
      k_from_dpq<<<ew_blocks((long)P2*128), 256, 0, stream>>>(DT, T0);
      k_ln<<<P2, 128, 0, stream>>>(T0, ln_out_g, ln_out_b, T2, 128);
      gemm(T2, p_out, T0, P2, 128, 128);
      gemm(ZN, g_out, T1, P2, 128, 128);
      k_gate_add<<<ew_blocks((long)P2*128), 256, 0, stream>>>(z, T0, T1, (long)P2*128);
    };
    tri_mult(TRI_OUT, 1);
    tri_mult(TRI_IN, 0);

    // ---- triangle attention (starting then ending) ----
    auto tri_attn = [&](int b, int starting){
      const float *lng = L[b+0], *lnb = L[b+1], *bias = L[b+2];
      const float *qw = L[b+3], *kw = L[b+4], *vw = L[b+5], *gw = L[b+6], *ow = L[b+7];
      const float* zin = z; const float* pmu = pm;
      if (!starting){
        k_transpose_ij<<<ew_blocks((long)P2*128), 256, 0, stream>>>(z, ZT, 128, (long)P2*128);
        k_transpose_ij<<<ew_blocks((long)P2), 256, 0, stream>>>(pm, PMT, 1, (long)P2);
        zin = ZT; pmu = PMT;
      }
      k_ln<<<P2, 128, 0, stream>>>(zin, lng, lnb, ZN, 128);
      k_gemm_small<<<ew_blocks((long)P2*4), 256, 0, stream>>>(ZN, bias, TBB, P2, 4, 128, 1);
      float* Qb = BIG0;
      float* Kb = BIG0 + (long)P2*128;
      float* Vb = BIG0 + (long)P2*256;
      float* Gb = BIG0 + (long)P2*384;
      gemm(ZN, qw, Qb, P2, 128, 128);
      gemm(ZN, kw, Kb, P2, 128, 128);
      gemm(ZN, vw, Vb, P2, 128, 128);
      gemm(ZN, gw, Gb, P2, 128, 128);
      dim3 ga(16, 4, 256);
      k_tri_attn<<<ga, 32, 0, stream>>>(Qb, Kb, Vb, TBB, pmu, T0);
      k_mul_sig<<<ew_blocks((long)P2*128), 256, 0, stream>>>(T0, Gb, T0, (long)P2*128);
      gemm(T0, ow, T1, P2, 128, 128);
      if (!starting){
        k_transpose_ij<<<ew_blocks((long)P2*128), 256, 0, stream>>>(T1, T2, 128, (long)P2*128);
        k_add<<<ew_blocks((long)P2*128), 256, 0, stream>>>(z, T2, (long)P2*128);
      } else {
        k_add<<<ew_blocks((long)P2*128), 256, 0, stream>>>(z, T1, (long)P2*128);
      }
    };
    tri_attn(ATT_START, 1);
    tri_attn(ATT_END, 0);

    // ---- SwiGLU transition ----
    auto transition = [&](float* x, int M, int D, int b, float* lnbuf, float* h1, float* h2, float* ob){
      const float *lng = L[b+0], *lnb = L[b+1], *fc1 = L[b+2], *fc2 = L[b+3], *fc3 = L[b+4];
      k_ln<<<M, 128, 0, stream>>>(x, lng, lnb, lnbuf, D);
      gemm(lnbuf, fc1, h1, M, 4*D, D);
      gemm(lnbuf, fc2, h2, M, 4*D, D);
      k_silu_mul<<<ew_blocks((long)M*4*D), 256, 0, stream>>>(h1, h2, h1, (long)M*4*D);
      gemm(h1, fc3, ob, M, D, 4*D);
      k_add<<<ew_blocks((long)M*D), 256, 0, stream>>>(x, ob, (long)M*D);
    };
    transition(z, P2, 128, TRANS_Z, ZN, BIG0, BIG1, T0);

    // ---- attention with pair bias on s ----
    k_ln<<<256, 128, 0, stream>>>(s, L[SLN_G], L[SLN_B], SN, TSc);
    gemm(SN, L[ATTN+0], SQ, 256, TSc, TSc);                       // wq
    k_add_bias<<<ew_blocks((long)256*TSc), 256, 0, stream>>>(SQ, L[ATTN+1], TSc, (long)256*TSc);
    gemm(SN, L[ATTN+2], SK, 256, TSc, TSc);                       // wk
    gemm(SN, L[ATTN+3], SV, 256, TSc, TSc);                       // wv
    k_ln<<<P2, 128, 0, stream>>>(z, L[ATTN+7], L[ATTN+8], ZN, 128); // zln
    k_gemm_small<<<ew_blocks((long)P2*16), 256, 0, stream>>>(ZN, L[ATTN+4], ZBB, P2, 16, 128, 0);
    dim3 gs(256, 16);
    k_s_attn<<<gs, 256, 0, stream>>>(SQ, SK, SV, ZBB, msk, SO);
    gemm(SN, L[ATTN+5], SG, 256, TSc, TSc);                       // wg
    k_mul_sig<<<ew_blocks((long)256*TSc), 256, 0, stream>>>(SO, SG, SO, (long)256*TSc);
    gemm(SO, L[ATTN+6], SO2, 256, TSc, TSc);                      // wo
    k_add<<<ew_blocks((long)256*TSc), 256, 0, stream>>>(s, SO2, (long)256*TSc);

    transition(s, 256, TSc, TRANS_S, SN, SH1, SH2, SO2);
  }
}
